// FedLEASEMoELayer_53274774340071
// MI455X (gfx1250) — matrix operand order repaired and checked by reading, not compiled, not run on hardware
//
#include <hip/hip_runtime.h>
#include <hip/hip_bf16.h>

// ---------------------------------------------------------------------------
// FedLEASE MoE layer for MI455X (gfx1250, wave32, WMMA bf16 16x16x32)
//   out = x @ W^T + b + SCALING * sum_e w_e * ((x @ A_e^T) @ B_e^T)
// Folded:  out = X_bf16 @ Wt + hw @ BcatT + b,  K = 4096 (+128 LoRA tail)
// Fast path: pre-converted bf16 operands + async global->LDS double buffering
// ---------------------------------------------------------------------------

#define M_TOK 8192   // B*T
#define DDIM  4096   // K
#define ODIM  4096   // N
#define NEXP  8
#define NRANK 16
#define NLOG  15     // 2E-1 router logits
#define K2    128    // E*R
#define SCALING 2.0f

typedef __attribute__((ext_vector_type(16))) __bf16 v16bf;
typedef __attribute__((ext_vector_type(8)))  __bf16 v8bf;
typedef __attribute__((ext_vector_type(8)))  float  v8f;

#define LDS_K 40   // 32 bf16 + 8 pad: 80B row pitch -> conflict-free, 16B aligned

// ---- helpers ---------------------------------------------------------------

static __device__ inline unsigned pack2bf(float a, float b) {
    union { __bf16 h[2]; unsigned u; } u;
    u.h[0] = (__bf16)a; u.h[1] = (__bf16)b;
    return u.u;
}

static __device__ inline v8f wmma_bf16(v16bf a, v16bf b, v8f c) {
    // D = A(16x32) * B(32x16) + C,  f32 accumulate
    return __builtin_amdgcn_wmma_f32_16x16x32_bf16(
        /*neg_a=*/false, a, /*neg_b=*/false, b,
        /*c_mod=*/(short)0, c, /*reuse_a=*/false, /*reuse_b=*/false);
}

// A fragment: 16x32 bf16, row = laneM, K split by lane-half:
//   lanes 0-15:  K in {0..7, 16..23};  lanes 16-31: K in {8..15, 24..31}
static __device__ inline v16bf ldsA(const __bf16* s, int row, int half) {
    union { v16bf v; v8bf h[2]; } u;
    const __bf16* p = s + row * LDS_K + half * 8;
    u.h[0] = *(const v8bf*)(p);
    u.h[1] = *(const v8bf*)(p + 16);
    return u.v;
}

// B fragment: 32x16 bf16, col = laneM, K split by lane-half (same as A):
//   lanes 0-15:  K in {0..7, 16..23};  lanes 16-31: K in {8..15, 24..31}
static __device__ inline v16bf ldsB(const __bf16* s, int col, int half) {
    union { v16bf v; v8bf h[2]; } u;
    const __bf16* p = s + col * LDS_K + half * 8;
    u.h[0] = *(const v8bf*)(p);
    u.h[1] = *(const v8bf*)(p + 16);
    return u.v;
}

// 8 WMMAs covering a 32(M) x 64(N) wave tile for one K=32 step
static __device__ inline void compute_step(const __bf16* sA, const __bf16* sB,
                                           int waveM, int waveN,
                                           int laneM, int half,
                                           v8f acc[2][4]) {
    v16bf af[2];
    v16bf bfr[4];
#pragma unroll
    for (int mi = 0; mi < 2; ++mi)
        af[mi] = ldsA(sA, waveM * 32 + mi * 16 + laneM, half);
#pragma unroll
    for (int ni = 0; ni < 4; ++ni)
        bfr[ni] = ldsB(sB, waveN * 64 + ni * 16 + laneM, half);
#pragma unroll
    for (int mi = 0; mi < 2; ++mi)
#pragma unroll
        for (int ni = 0; ni < 4; ++ni)
            acc[mi][ni] = wmma_bf16(af[mi], bfr[ni], acc[mi][ni]);
}

// ---- CDNA5 async global->LDS copy (ASYNCcnt) -------------------------------
// the LDS destination is a byte offset held in a VGPR; generic LDS pointer low 32 bits are the
// wave-relative LDS byte offset (aperture in high bits), so truncate.
static __device__ inline void async_ld16(const __bf16* lds_ptr, const __bf16* g) {
    unsigned off = (unsigned)(uintptr_t)lds_ptr;
    asm volatile("global_load_async_to_lds_b128 %0, %1, off"
                 :: "v"(off), "v"(g) : "memory");
}
static __device__ inline void s_wait_async4() {
    asm volatile("s_wait_asynccnt 0x4" ::: "memory");
}
static __device__ inline void s_wait_async0() {
    asm volatile("s_wait_asynccnt 0x0" ::: "memory");
}

// Issue one 128x32 bf16 tile pair into LDS: 4 async b128 per thread/wave.
static __device__ inline void issue_tile(const __bf16* gA, int ldA,
                                         const __bf16* gB, int ldB,
                                         __bf16* sa, __bf16* sb,
                                         int row, int kq) {
    const __bf16* pa = gA + (size_t)row * ldA + kq;
    const __bf16* pb = gB + (size_t)row * ldB + kq;
    async_ld16(&sa[row * LDS_K + kq], pa);
    async_ld16(&sa[(row + 64) * LDS_K + kq], pa + (size_t)64 * ldA);
    async_ld16(&sb[row * LDS_K + kq], pb);
    async_ld16(&sb[(row + 64) * LDS_K + kq], pb + (size_t)64 * ldB);
}

// ---- kernel 1: router -> per-expert combine weight w[t][e] * SCALING -------
__global__ __launch_bounds__(256)
void router_kernel(const float* __restrict__ x,
                   const float* __restrict__ rW,
                   const float* __restrict__ rb,
                   const int*   __restrict__ expert_map,
                   float* __restrict__ wbuf) {
    const int lane  = threadIdx.x & 31;
    const int token = blockIdx.x * 8 + (threadIdx.x >> 5);

    float acc[NLOG];
#pragma unroll
    for (int n = 0; n < NLOG; ++n) acc[n] = 0.0f;

    const float* xp = x + (size_t)token * DDIM;
    for (int d = lane * 4; d < DDIM; d += 32 * 4) {
        float4 xv = *(const float4*)(xp + d);
#pragma unroll
        for (int n = 0; n < NLOG; ++n) {
            float4 wv = *(const float4*)(rW + (size_t)n * DDIM + d);
            acc[n] += xv.x * wv.x + xv.y * wv.y + xv.z * wv.z + xv.w * wv.w;
        }
    }
#pragma unroll
    for (int n = 0; n < NLOG; ++n) {
#pragma unroll
        for (int off = 16; off > 0; off >>= 1)
            acc[n] += __shfl_xor(acc[n], off, 32);
    }

    float mx = -1e30f;
#pragma unroll
    for (int n = 0; n < NLOG; ++n) { acc[n] += rb[n]; mx = fmaxf(mx, acc[n]); }
    float p[NLOG]; float sum = 0.0f;
#pragma unroll
    for (int n = 0; n < NLOG; ++n) { p[n] = __expf(acc[n] - mx); sum += p[n]; }
    const float inv = 1.0f / sum;
#pragma unroll
    for (int n = 0; n < NLOG; ++n) p[n] *= inv;

    float w[NEXP];
#pragma unroll
    for (int e = 0; e < NEXP; ++e) w[e] = 0.0f;
    bool used[NLOG];
#pragma unroll
    for (int n = 0; n < NLOG; ++n) used[n] = false;
    for (int s = 0; s < NEXP; ++s) {            // top-8 of 15, first wins
        int   bi = -1;
        float bv = -1.0f;
        for (int n = 0; n < NLOG; ++n)
            if (!used[n] && p[n] > bv) { bv = p[n]; bi = n; }
        used[bi] = true;
        w[expert_map[bi]] += bv;
    }
    if (lane == 0) {
#pragma unroll
        for (int e = 0; e < NEXP; ++e)
            wbuf[(size_t)token * NEXP + e] = w[e] * SCALING;
    }
}

// ---- kernel 2: BcatT[o][j=e*16+r] = bf16(B[e][o][r])  ----------------------
__global__ __launch_bounds__(256)
void bcat_kernel(const float* __restrict__ Bm, __bf16* __restrict__ bcatT) {
    int idx = blockIdx.x * 256 + threadIdx.x;   // o*128 + j
    int o = idx >> 7;
    int j = idx & (K2 - 1);
    int e = j >> 4;
    int r = j & (NRANK - 1);
    bcatT[idx] = (__bf16)Bm[((size_t)e * ODIM + o) * NRANK + r];
}

// ---- kernel 2b: fp32 -> bf16 bulk convert (for async fast path) ------------
__global__ __launch_bounds__(256)
void cvt_bf16_kernel(const float* __restrict__ src, __bf16* __restrict__ dst) {
    size_t i = ((size_t)blockIdx.x * 256 + threadIdx.x) * 4;
    float4 v = *(const float4*)(src + i);
    uint2 p{pack2bf(v.x, v.y), pack2bf(v.z, v.w)};
    *(uint2*)(dst + i) = p;
}

// ---- kernel 3: hw[t][j] = w[t][j/16] * (x @ A2^T)  (WMMA, N = 128) ---------
__global__ __launch_bounds__(256)
void h_gemm_kernel(const float* __restrict__ x,
                   const float* __restrict__ A2,     // [128][4096] = A[e][r][:]
                   const float* __restrict__ wbuf,
                   __bf16* __restrict__ hw) {
    __shared__ __bf16 sA[128 * LDS_K];
    __shared__ __bf16 sB[128 * LDS_K];

    const int tid   = threadIdx.x;
    const int lane  = tid & 31, wave = tid >> 5;
    const int half  = lane >> 4, laneM = lane & 15;
    const int waveM = wave & 3,  waveN = wave >> 2;
    const int mBase = blockIdx.y * 128;

    v8f acc[2][4] = {};

    for (int k0 = 0; k0 < DDIM; k0 += 32) {
#pragma unroll
        for (int it = 0; it < 4; ++it) {
            int idx = tid + it * 256;           // 1024 float4 chunks / tile
            int row = idx >> 3;
            int kq  = (idx & 7) << 2;
            float4 xv = *(const float4*)(x  + (size_t)(mBase + row) * DDIM + k0 + kq);
            float4 av = *(const float4*)(A2 + (size_t)row * DDIM + k0 + kq);
            uint2 xs{pack2bf(xv.x, xv.y), pack2bf(xv.z, xv.w)};
            uint2 as{pack2bf(av.x, av.y), pack2bf(av.z, av.w)};
            *(uint2*)(&sA[row * LDS_K + kq]) = xs;
            *(uint2*)(&sB[row * LDS_K + kq]) = as;
        }
        __syncthreads();
        compute_step(sA, sB, waveM, waveN, laneM, half, acc);
        __syncthreads();
    }

#pragma unroll
    for (int mi = 0; mi < 2; ++mi)
#pragma unroll
        for (int ni = 0; ni < 4; ++ni) {
            int col = waveN * 64 + ni * 16 + laneM;         // j
            int rb  = mBase + waveM * 32 + mi * 16 + half * 8;
#pragma unroll
            for (int i = 0; i < 8; ++i) {
                int row = rb + i;                           // token
                float ws = wbuf[(size_t)row * NEXP + (col >> 4)];
                hw[(size_t)row * K2 + col] = (__bf16)(acc[mi][ni][i] * ws);
            }
        }
}

// ---- kernel 4a: FAST PATH — async double-buffered bf16 GEMM ----------------
__global__ __launch_bounds__(256)
void main_gemm_async_kernel(const __bf16* __restrict__ xbf,   // [8192][4096]
                            const __bf16* __restrict__ wbf,   // [4096][4096]
                            const float*  __restrict__ bias,
                            const __bf16* __restrict__ hw,    // [8192][128]
                            const __bf16* __restrict__ bcatT, // [4096][128]
                            float* __restrict__ out) {
    __shared__ __bf16 sA[2][128 * LDS_K];
    __shared__ __bf16 sB[2][128 * LDS_K];

    const int tid   = threadIdx.x;
    const int lane  = tid & 31, wave = tid >> 5;
    const int half  = lane >> 4, laneM = lane & 15;
    const int waveM = wave & 3,  waveN = wave >> 2;
    const int mBase = blockIdx.y * 128;
    const int nBase = blockIdx.x * 128;

    // staging coords: 512 16B chunks per 128x32 tile; this thread owns 2/op
    const int row = tid >> 2;
    const int kq  = (tid & 3) << 3;

    const __bf16* gX = xbf   + (size_t)mBase * DDIM;
    const __bf16* gW = wbf   + (size_t)nBase * DDIM;
    const __bf16* gH = hw    + (size_t)mBase * K2;
    const __bf16* gC = bcatT + (size_t)nBase * K2;

    v8f acc[2][4] = {};

    const int nMain = DDIM / 32;        // 128
    const int nTot  = nMain + K2 / 32;  // 132 (LoRA tail fused)

    issue_tile(gX, DDIM, gW, DDIM, sA[0], sB[0], row, kq);   // prime buf 0

    for (int t = 0; t < nTot; ++t) {
        const int cur = t & 1;
        if (t + 1 < nTot) {
            const int u = t + 1;
            if (u < nMain)
                issue_tile(gX + u * 32, DDIM, gW + u * 32, DDIM,
                           sA[cur ^ 1], sB[cur ^ 1], row, kq);
            else
                issue_tile(gH + (u - nMain) * 32, K2, gC + (u - nMain) * 32, K2,
                           sA[cur ^ 1], sB[cur ^ 1], row, kq);
            s_wait_async4();    // current tile landed; next still in flight
        } else {
            s_wait_async0();
        }
        __syncthreads();
        compute_step(sA[cur], sB[cur], waveM, waveN, laneM, half, acc);
        __syncthreads();        // all reads done before buffer is re-filled
    }

#pragma unroll
    for (int mi = 0; mi < 2; ++mi)
#pragma unroll
        for (int ni = 0; ni < 4; ++ni) {
            int col = nBase + waveN * 64 + ni * 16 + laneM;
            int rb  = mBase + waveM * 32 + mi * 16 + half * 8;
            float bs = bias[col];
#pragma unroll
            for (int i = 0; i < 8; ++i)
                out[(size_t)(rb + i) * ODIM + col] = acc[mi][ni][i] + bs;
        }
}

// ---- kernel 4b: FALLBACK — in-kernel fp32->bf16 conversion GEMM ------------
__global__ __launch_bounds__(256)
void main_gemm_sync_kernel(const float* __restrict__ x,
                           const float* __restrict__ Wm,
                           const float* __restrict__ bias,
                           const __bf16* __restrict__ hw,
                           const __bf16* __restrict__ bcatT,
                           float* __restrict__ out) {
    __shared__ __bf16 sA[128 * LDS_K];
    __shared__ __bf16 sB[128 * LDS_K];

    const int tid   = threadIdx.x;
    const int lane  = tid & 31, wave = tid >> 5;
    const int half  = lane >> 4, laneM = lane & 15;
    const int waveM = wave & 3,  waveN = wave >> 2;
    const int mBase = blockIdx.y * 128;
    const int nBase = blockIdx.x * 128;

    v8f acc[2][4] = {};

    for (int k0 = 0; k0 < DDIM; k0 += 32) {
#pragma unroll
        for (int it = 0; it < 4; ++it) {
            int idx = tid + it * 256;
            int row = idx >> 3;
            int kq  = (idx & 7) << 2;
            const float* xp = x  + (size_t)(mBase + row) * DDIM + k0 + kq;
            const float* wp = Wm + (size_t)(nBase + row) * DDIM + k0 + kq;
            if (k0 + 32 < DDIM) {                   // global_prefetch next tile
                __builtin_prefetch(xp + 32, 0, 3);
                __builtin_prefetch(wp + 32, 0, 3);
            }
            float4 xv = *(const float4*)xp;
            float4 wv = *(const float4*)wp;
            uint2 xs{pack2bf(xv.x, xv.y), pack2bf(xv.z, xv.w)};
            uint2 ws{pack2bf(wv.x, wv.y), pack2bf(wv.z, wv.w)};
            *(uint2*)(&sA[row * LDS_K + kq]) = xs;
            *(uint2*)(&sB[row * LDS_K + kq]) = ws;
        }
        __syncthreads();
        compute_step(sA, sB, waveM, waveN, laneM, half, acc);
        __syncthreads();
    }

    for (int k0 = 0; k0 < K2; k0 += 32) {
#pragma unroll
        for (int it = 0; it < 2; ++it) {
            int idx = tid + it * 256;
            int row = idx >> 2;
            int kq  = (idx & 3) << 3;
            uint4 hv = *(const uint4*)(hw    + (size_t)(mBase + row) * K2 + k0 + kq);
            uint4 bv = *(const uint4*)(bcatT + (size_t)(nBase + row) * K2 + k0 + kq);
            *(uint4*)(&sA[row * LDS_K + kq]) = hv;
            *(uint4*)(&sB[row * LDS_K + kq]) = bv;
        }
        __syncthreads();
        compute_step(sA, sB, waveM, waveN, laneM, half, acc);
        __syncthreads();
    }

#pragma unroll
    for (int mi = 0; mi < 2; ++mi)
#pragma unroll
        for (int ni = 0; ni < 4; ++ni) {
            int col = nBase + waveN * 64 + ni * 16 + laneM;
            int rb  = mBase + waveM * 32 + mi * 16 + half * 8;
            float bs = bias[col];
#pragma unroll
            for (int i = 0; i < 8; ++i)
                out[(size_t)(rb + i) * ODIM + col] = acc[mi][ni][i] + bs;
        }
}

// ---------------------------------------------------------------------------
extern "C" void kernel_launch(void* const* d_in, const int* in_sizes, int n_in,
                              void* d_out, int out_size, void* d_ws, size_t ws_size,
                              hipStream_t stream) {
    (void)in_sizes; (void)n_in; (void)out_size;
    const float* x        = (const float*)d_in[0];
    const float* base_W   = (const float*)d_in[1];
    const float* base_b   = (const float*)d_in[2];
    const float* router_W = (const float*)d_in[3];
    const float* router_b = (const float*)d_in[4];
    const float* A        = (const float*)d_in[5];   // [E][R][D] == A2[128][4096]
    const float* Bm       = (const float*)d_in[6];   // [E][O][R]
    const int*   emap     = (const int*)d_in[7];     // [15]
    float* out = (float*)d_out;

    // workspace layout
    char* ws = (char*)d_ws;
    float*  wbuf  = (float*)ws;                                   // 256 KB
    __bf16* hw    = (__bf16*)(ws + (256ull << 10));               // 2 MB
    __bf16* bcatT = (__bf16*)(ws + (256ull << 10) + (2ull << 20));// 1 MB
    const size_t OFF_XBF = 4ull << 20;
    const size_t OFF_WBF = OFF_XBF + (64ull << 20);
    const size_t NEED    = OFF_WBF + (32ull << 20);               // ~100 MB

    router_kernel<<<M_TOK / 8, 256, 0, stream>>>(x, router_W, router_b, emap, wbuf);
    bcat_kernel<<<(ODIM * K2) / 256, 256, 0, stream>>>(Bm, bcatT);
    h_gemm_kernel<<<dim3(1, M_TOK / 128), 256, 0, stream>>>(x, A, wbuf, hw);

    if (ws_size >= NEED) {
        // fast path: bulk fp32->bf16 converts, then async double-buffered GEMM
        __bf16* xbf = (__bf16*)(ws + OFF_XBF);
        __bf16* wbf = (__bf16*)(ws + OFF_WBF);
        cvt_bf16_kernel<<<(M_TOK * (size_t)DDIM) / 4 / 256, 256, 0, stream>>>(x, xbf);
        cvt_bf16_kernel<<<(ODIM * (size_t)DDIM) / 4 / 256, 256, 0, stream>>>(base_W, wbf);
        main_gemm_async_kernel<<<dim3(ODIM / 128, M_TOK / 128), 256, 0, stream>>>(
            xbf, wbf, base_b, hw, bcatT, out);
    } else {
        main_gemm_sync_kernel<<<dim3(ODIM / 128, M_TOK / 128), 256, 0, stream>>>(
            x, base_W, base_b, hw, bcatT, out);
    }
}
